// joint_model_19447611916620
// MI455X (gfx1250) — matrix operand order repaired and checked by reading, not compiled, not run on hardware
//
#include <hip/hip_runtime.h>

// ---------------------------------------------------------------------------
// Problem constants (match reference)
// ---------------------------------------------------------------------------
#define Dd      256
#define Bsz     131072
#define Lw      8
#define S1c     128
#define S2c     128
#define NCc     10
#define NWG     1024          // flow workgroups, 128 batch rows each
#define MT      128           // rows per workgroup tile
#define XSTR    264           // f32 LDS row stride (pad 8 floats)
#define HSTR    264           // bf16 LDS row stride (pad 8 elems)
#define X1STR   136           // bf16 x1 buffer row stride (272B -> 4-bank stagger)
#define CLAMP_C   1.9f
#define GCONST_C  (-0.9189385332046727f)
#define BN_EPS_C  1e-5f

typedef __attribute__((ext_vector_type(16))) __bf16 v16bf;
typedef __attribute__((ext_vector_type(8)))  __bf16 v8bf;
typedef __attribute__((ext_vector_type(4)))  __bf16 v4bf;
typedef __attribute__((ext_vector_type(8)))  float  v8f;

// D = A(16x32 bf16) x B(32x16 bf16) + C(16x16 f32)
__device__ __forceinline__ v8f wmma_bf16(v16bf a, v16bf b, v8f c) {
    return __builtin_amdgcn_wmma_f32_16x16x32_bf16(
        /*neg_a=*/false, a, /*neg_b=*/false, b,
        /*c_mod=*/(short)0, c, /*reuse_a=*/false, /*reuse_b=*/false);
}

// B fragment: weights stored [N][K] bf16 row-major (K mult of 32).
// B lane order chosen here (k=16h+s, same as loadA_bf): lanes 0-15 -> N=n0..n0+15, K chunk {2j,2j+1};
// lanes 16-31 same N, K offset +16.  One contiguous 32B load per lane.
__device__ __forceinline__ v16bf loadB(const __bf16* __restrict__ Wt, int K,
                                       int n0, int ks, int lo, int hs) {
    const __bf16* p = Wt + (size_t)(n0 + lo) * K + ks * 32 + hs * 16;
    return *(const v16bf*)p;
}

// A fragment from bf16 LDS rows (two 16B ds loads).
// A 16x32, order chosen here (k=16h+s, as loadB): lanes 0-15 hold K {kb..kb+15};
// lanes 16-31 hold K {kb+16..kb+31}.
__device__ __forceinline__ v16bf loadA_bf(const __bf16* buf, int stride, int m0,
                                          int kb, int lo, int hs) {
    const __bf16* p = buf + (size_t)(m0 + lo) * stride + kb + hs * 16;
    v8bf x = *(const v8bf*)(p);
    v8bf y = *(const v8bf*)(p + 8);
    v16bf A;
#pragma unroll
    for (int i = 0; i < 8; ++i) { A[i] = x[i]; A[8 + i] = y[i]; }
    return A;
}

// ---------------------------------------------------------------------------
// Kernel 0a: transpose + convert weights f32 -> bf16 [N][K] in workspace
// ---------------------------------------------------------------------------
__global__ void kConvert(const float* __restrict__ W1, const float* __restrict__ W2,
                         const float* __restrict__ P,  const float* __restrict__ Wh1,
                         __bf16* __restrict__ W1t, __bf16* __restrict__ W2t,
                         __bf16* __restrict__ Pt,  __bf16* __restrict__ Wh1t) {
    int i = blockIdx.x * 256 + threadIdx.x;
    const int N1 = Lw * Dd * S1c;     // 262144  W1t [l][n:256][k:128]
    const int N2 = Lw * Dd * Dd;      // 524288  W2t [l][n:256][k:256]
    if (i < N1) {
        int l = i / (Dd * S1c), r = i % (Dd * S1c), n = r / S1c, k = r % S1c;
        W1t[i] = (__bf16)W1[(size_t)l * S1c * Dd + (size_t)k * Dd + n];
    } else if (i < N1 + N2) {
        int j = i - N1;
        int l = j / (Dd * Dd), r = j % (Dd * Dd), n = r / Dd, k = r % Dd;
        W2t[j] = (__bf16)W2[(size_t)l * Dd * Dd + (size_t)k * Dd + n];
    } else if (i < N1 + 2 * N2) {
        int j = i - N1 - N2;          // P used as P^T: B elem (K=k,N=n)=P[n][k]
        Pt[j] = (__bf16)P[j];
    } else if (i < N1 + 2 * N2 + Dd * Dd) {
        int j = i - N1 - 2 * N2;
        int n = j / Dd, k = j % Dd;
        Wh1t[j] = (__bf16)Wh1[(size_t)k * Dd + n];
    }
}

// ---------------------------------------------------------------------------
// Kernel 0b: scale = 0.2*softplus(0.5*g); per-layer sum(log(scale))
// ---------------------------------------------------------------------------
__global__ void kScale(const float* __restrict__ gs, float* __restrict__ scaleV,
                       float* __restrict__ ldconst) {
    __shared__ float red[256];
    int l = blockIdx.x, d = threadIdx.x;
    float x  = 0.5f * gs[l * Dd + d];
    float sp = (x > 20.f) ? x : log1pf(expf(x));
    float sc = 0.2f * sp;
    scaleV[l * Dd + d] = sc;
    red[d] = logf(sc);
    __syncthreads();
    for (int s = 128; s > 0; s >>= 1) {
        if (d < s) red[d] += red[d + s];
        __syncthreads();
    }
    if (d == 0) ldconst[l] = red[0];
}

// ---------------------------------------------------------------------------
// Kernel 1: the flow (8 coupling layers) + log_px + head matmul h=z@Wh1+bh1
// 256 threads = 8 wave32s; 128-row activation tile resident in LDS.
// ---------------------------------------------------------------------------
__global__ __launch_bounds__(256, 1)
void kFlow(const float* __restrict__ feat,
           const float* __restrict__ b1g, const float* __restrict__ b2g,
           const float* __restrict__ gog, const float* __restrict__ bh1g,
           const __bf16* __restrict__ W1t, const __bf16* __restrict__ W2t,
           const __bf16* __restrict__ Pt,  const __bf16* __restrict__ Wh1t,
           const float* __restrict__ scaleV, const float* __restrict__ ldconst,
           float* __restrict__ Hg, float* __restrict__ partial,
           float* __restrict__ out_logpx) {
    __shared__ float  Xs[MT * XSTR];    // f32 activation state   (132KB)
    __shared__ __bf16 Hb[MT * HSTR];    // bf16 matmul-A staging  (66KB)
    __shared__ __bf16 X1b[MT * X1STR];  // bf16 copy of x1        (34KB)
    __shared__ float  ldp[8 * MT];      // per-wave logdet row partials

    const int tid  = threadIdx.x;
    const int w    = tid >> 5;          // wave id 0..7
    const int lane = tid & 31;
    const int lo   = lane & 15;
    const int hs   = lane >> 4;
    const int rowBase = blockIdx.x * MT;

    for (int i = tid; i < 8 * MT; i += 256) ldp[i] = 0.f;
    for (int i = tid; i < MT * (Dd / 4); i += 256) {
        int m = i >> 6, c = (i & 63) * 4;
        float4 v = *(const float4*)&feat[(size_t)(rowBase + m) * Dd + c];
        *(float4*)&Xs[m * XSTR + c] = v;
        if (c < S1c) {                   // bf16 x1 copy for layer-0 matmul1
            v4bf o;
            o[0] = (__bf16)v.x; o[1] = (__bf16)v.y;
            o[2] = (__bf16)v.z; o[3] = (__bf16)v.w;
            *(v4bf*)&X1b[m * X1STR + c] = o;
        }
    }
    __syncthreads();

#pragma unroll 1
    for (int l = 0; l < Lw; ++l) {
        const __bf16* w1 = W1t + (size_t)l * Dd * S1c;  // [256][128]
        const __bf16* w2 = W2t + (size_t)l * Dd * Dd;   // [256][256]
        const __bf16* pp = Pt  + (size_t)l * Dd * Dd;   // [256][256]
        if (lane == 0) {  // warm L2/L0 for the weights of this layer
            __builtin_prefetch(w2, 0, 1);
            __builtin_prefetch(pp, 0, 1);
        }

        // ---- matmul1: hidden = relu(x1 @ W1 + b1)  -> Hb (bf16), K=128
        {
            v16bf Bf[2][4];
            float bias[2];
#pragma unroll
            for (int h = 0; h < 2; ++h) {
                int n0 = 16 * w + 128 * h;
                bias[h] = b1g[l * Dd + n0 + lo];
#pragma unroll
                for (int ks = 0; ks < 4; ++ks) Bf[h][ks] = loadB(w1, S1c, n0, ks, lo, hs);
            }
            for (int mt = 0; mt < 8; ++mt) {
                v8f acc0 = {}, acc1 = {};
#pragma unroll
                for (int ks = 0; ks < 4; ++ks) {
                    v16bf Af = loadA_bf(X1b, X1STR, mt * 16, ks * 32, lo, hs);
                    acc0 = wmma_bf16(Af, Bf[0][ks], acc0);
                    acc1 = wmma_bf16(Af, Bf[1][ks], acc1);
                }
#pragma unroll
                for (int r = 0; r < 8; ++r) {
                    int m = mt * 16 + r + 8 * hs;
                    float v0 = acc0[r] + bias[0]; v0 = v0 > 0.f ? v0 : 0.f;
                    float v1 = acc1[r] + bias[1]; v1 = v1 > 0.f ? v1 : 0.f;
                    Hb[m * HSTR + 16 * w + lo]       = (__bf16)v0;
                    Hb[m * HSTR + 128 + 16 * w + lo] = (__bf16)v1;
                }
            }
        }
        __syncthreads();

        // ---- matmul2: a = 0.1*(hidden @ W2 + b2); coupling update, K=256
        // wave w owns paired strips: s-part cols [16w,16w+16), t-part +128.
        {
            v16bf BS[8], BT[8];
#pragma unroll
            for (int ks = 0; ks < 8; ++ks) {
                BS[ks] = loadB(w2, Dd, 16 * w,       ks, lo, hs);
                BT[ks] = loadB(w2, Dd, 128 + 16 * w, ks, lo, hs);
            }
            float b2S = b2g[l * Dd + 16 * w + lo];
            float b2T = b2g[l * Dd + 128 + 16 * w + lo];
            for (int mt = 0; mt < 8; ++mt) {
                v8f aS = {}, aT = {};
#pragma unroll
                for (int ks = 0; ks < 8; ++ks) {
                    v16bf Af = loadA_bf(Hb, HSTR, mt * 16, ks * 32, lo, hs);
                    aS = wmma_bf16(Af, BS[ks], aS);
                    aT = wmma_bf16(Af, BT[ks], aT);
                }
#pragma unroll
                for (int r = 0; r < 8; ++r) {
                    int   m  = mt * 16 + r + 8 * hs;
                    float av = 0.1f * (aS[r] + b2S);
                    float tv = 0.1f * (aT[r] + b2T);
                    float s  = CLAMP_C * tanhf(av);
                    int   c2 = 128 + 16 * w + lo;    // exclusive column
                    float x2 = Xs[m * XSTR + c2];
                    Xs[m * XSTR + c2] = x2 * __expf(s) + tv;
                    // row-sum of s over the wave's 16 columns (per lane-half)
                    float rs = s;
                    rs += __shfl_xor(rs, 1); rs += __shfl_xor(rs, 2);
                    rs += __shfl_xor(rs, 4); rs += __shfl_xor(rs, 8);
                    if (lo == 0) ldp[w * MT + m] += rs;
                }
            }
        }
        __syncthreads();

        // ---- y = x * scale + o  -> Hb (bf16), 16B vector stores
        for (int i = tid; i < MT * (Dd / 8); i += 256) {
            int m = i >> 5, c = (i & 31) * 8;
            const float* px = &Xs[m * XSTR + c];
            const float* psc = &scaleV[l * Dd + c];
            const float* po  = &gog[l * Dd + c];
            v8bf o;
#pragma unroll
            for (int j = 0; j < 8; ++j)
                o[j] = (__bf16)(px[j] * psc[j] + po[j]);
            *(v8bf*)&Hb[m * HSTR + c] = o;
        }
        __syncthreads();

        // ---- matmul3: xo = y @ P^T -> Xs (f32); strip-0 also -> X1b (next x1)
        {
            v16bf BP[2][8];
#pragma unroll
            for (int h = 0; h < 2; ++h)
#pragma unroll
                for (int ks = 0; ks < 8; ++ks)
                    BP[h][ks] = loadB(pp, Dd, 16 * w + 128 * h, ks, lo, hs);
            for (int mt = 0; mt < 8; ++mt) {
                v8f a0 = {}, a1 = {};
#pragma unroll
                for (int ks = 0; ks < 8; ++ks) {
                    v16bf Af = loadA_bf(Hb, HSTR, mt * 16, ks * 32, lo, hs);
                    a0 = wmma_bf16(Af, BP[0][ks], a0);
                    a1 = wmma_bf16(Af, BP[1][ks], a1);
                }
#pragma unroll
                for (int r = 0; r < 8; ++r) {
                    int m = mt * 16 + r + 8 * hs;
                    Xs[m * XSTR + 16 * w + lo]       = a0[r];
                    Xs[m * XSTR + 128 + 16 * w + lo] = a1[r];
                    // cols 16w+lo < 128 are exactly next layer's x1
                    X1b[m * X1STR + 16 * w + lo]     = (__bf16)a0[r];
                }
            }
        }
        __syncthreads();
    }

    // ---- log_px (Xs now holds z)
    if (tid < MT) {
        float ss = 0.f;
        for (int c = 0; c < Dd; c += 4) {
            float4 v = *(const float4*)&Xs[tid * XSTR + c];
            ss += v.x * v.x + v.y * v.y + v.z * v.z + v.w * v.w;
        }
        float ld = 0.f;
#pragma unroll
        for (int ww = 0; ww < 8; ++ww) ld += ldp[ww * MT + tid];
#pragma unroll
        for (int l = 0; l < Lw; ++l) ld += ldconst[l];
        float logp = (float)Dd * GCONST_C - 0.5f * ss + ld;
        out_logpx[rowBase + tid] = logp / (float)Dd;
    }
    // z -> bf16 staging for head matmul (16B vector stores)
    for (int i = tid; i < MT * (Dd / 8); i += 256) {
        int m = i >> 5, c = (i & 31) * 8;
        const float* px = &Xs[m * XSTR + c];
        v8bf o;
#pragma unroll
        for (int j = 0; j < 8; ++j) o[j] = (__bf16)px[j];
        *(v8bf*)&Hb[m * HSTR + c] = o;
    }
    __syncthreads();

    // ---- head: h = z @ Wh1 + bh1 -> Hg, with per-WG column sums for BN
    {
        v16bf Bh[2][8];
        float bias[2];
#pragma unroll
        for (int h = 0; h < 2; ++h) {
            int n0 = 16 * w + 128 * h;
            bias[h] = bh1g[n0 + lo];
#pragma unroll
            for (int ks = 0; ks < 8; ++ks) Bh[h][ks] = loadB(Wh1t, Dd, n0, ks, lo, hs);
        }
        float cs0 = 0.f, cq0 = 0.f, cs1 = 0.f, cq1 = 0.f;
        for (int mt = 0; mt < 8; ++mt) {
            v8f a0 = {}, a1 = {};
#pragma unroll
            for (int ks = 0; ks < 8; ++ks) {
                v16bf Af = loadA_bf(Hb, HSTR, mt * 16, ks * 32, lo, hs);
                a0 = wmma_bf16(Af, Bh[0][ks], a0);
                a1 = wmma_bf16(Af, Bh[1][ks], a1);
            }
#pragma unroll
            for (int r = 0; r < 8; ++r) {
                int   m  = mt * 16 + r + 8 * hs;
                float h0 = a0[r] + bias[0];
                float h1 = a1[r] + bias[1];
                Hg[(size_t)(rowBase + m) * Dd + 16 * w + lo]       = h0;
                Hg[(size_t)(rowBase + m) * Dd + 128 + 16 * w + lo] = h1;
                cs0 += h0; cq0 += h0 * h0;
                cs1 += h1; cq1 += h1 * h1;
            }
        }
        cs0 += __shfl_xor(cs0, 16); cq0 += __shfl_xor(cq0, 16);
        cs1 += __shfl_xor(cs1, 16); cq1 += __shfl_xor(cq1, 16);
        if (lane < 16) {   // deterministic per-WG slots, no float atomics
            float* pb = partial + (size_t)blockIdx.x * 512;
            pb[16 * w + lo]            = cs0;
            pb[Dd + 16 * w + lo]       = cq0;
            pb[128 + 16 * w + lo]      = cs1;
            pb[Dd + 128 + 16 * w + lo] = cq1;
        }
    }
}

// ---------------------------------------------------------------------------
// Kernel 2: reduce BN partials (fixed order) -> fused coefficients A, B
// ---------------------------------------------------------------------------
__global__ void kStats(const float* __restrict__ partial,
                       const float* __restrict__ gamma, const float* __restrict__ beta,
                       float* __restrict__ stats) {
    int c = threadIdx.x;  // 256
    float s = 0.f, s2 = 0.f;
    for (int g = 0; g < NWG; ++g) {
        const float* pb = partial + (size_t)g * 512;
        s  += pb[c];
        s2 += pb[Dd + c];
    }
    float mu  = s / (float)Bsz;
    float var = s2 / (float)Bsz - mu * mu;
    float rs  = rsqrtf(var + BN_EPS_C);
    float A   = rs * gamma[c];
    stats[c]      = A;                 // h_norm = h*A + B
    stats[Dd + c] = beta[c] - mu * A;
}

// ---------------------------------------------------------------------------
// Kernel 3: logits = relu(h*A + B) @ Wh2 + bh2
// ---------------------------------------------------------------------------
__global__ __launch_bounds__(256)
void kLogits(const float* __restrict__ Hg, const float* __restrict__ stats,
             const float* __restrict__ Wh2, const float* __restrict__ bh2,
             float* __restrict__ out) {
    __shared__ float sA[Dd], sB[Dd], sW[Dd * NCc], sb[NCc];
    int tid = threadIdx.x;
    sA[tid] = stats[tid];
    sB[tid] = stats[Dd + tid];
    for (int i = tid; i < Dd * NCc; i += 256) sW[i] = Wh2[i];
    if (tid < NCc) sb[tid] = bh2[tid];
    __syncthreads();
    size_t row = (size_t)blockIdx.x * 256 + tid;
    float acc[NCc];
#pragma unroll
    for (int k = 0; k < NCc; ++k) acc[k] = sb[k];
    const float* hrow = Hg + row * Dd;
    for (int c = 0; c < Dd; c += 4) {
        float4 hv = *(const float4*)(hrow + c);
        float h0 = fmaf(hv.x, sA[c],     sB[c]);     h0 = h0 > 0.f ? h0 : 0.f;
        float h1 = fmaf(hv.y, sA[c + 1], sB[c + 1]); h1 = h1 > 0.f ? h1 : 0.f;
        float h2 = fmaf(hv.z, sA[c + 2], sB[c + 2]); h2 = h2 > 0.f ? h2 : 0.f;
        float h3 = fmaf(hv.w, sA[c + 3], sB[c + 3]); h3 = h3 > 0.f ? h3 : 0.f;
#pragma unroll
        for (int k = 0; k < NCc; ++k) {
            acc[k] = fmaf(h0, sW[(c)     * NCc + k], acc[k]);
            acc[k] = fmaf(h1, sW[(c + 1) * NCc + k], acc[k]);
            acc[k] = fmaf(h2, sW[(c + 2) * NCc + k], acc[k]);
            acc[k] = fmaf(h3, sW[(c + 3) * NCc + k], acc[k]);
        }
    }
    float* o = out + (size_t)Bsz + row * NCc;
#pragma unroll
    for (int k = 0; k < NCc; ++k) o[k] = acc[k];
}

// ---------------------------------------------------------------------------
// Host launcher
// ---------------------------------------------------------------------------
extern "C" void kernel_launch(void* const* d_in, const int* in_sizes, int n_in,
                              void* d_out, int out_size, void* d_ws, size_t ws_size,
                              hipStream_t stream) {
    (void)in_sizes; (void)n_in; (void)out_size; (void)ws_size;
    const float* feat  = (const float*)d_in[0];
    const float* W1    = (const float*)d_in[1];
    const float* b1    = (const float*)d_in[2];
    const float* W2    = (const float*)d_in[3];
    const float* b2    = (const float*)d_in[4];
    const float* gs    = (const float*)d_in[5];
    const float* go    = (const float*)d_in[6];
    const float* P     = (const float*)d_in[7];
    const float* Wh1   = (const float*)d_in[8];
    const float* bh1   = (const float*)d_in[9];
    const float* gamma = (const float*)d_in[10];
    const float* beta  = (const float*)d_in[11];
    const float* Wh2   = (const float*)d_in[12];
    const float* bh2   = (const float*)d_in[13];
    float* out = (float*)d_out;

    // Workspace layout (≈133 MiB): h buffer, BN partials, stats, scale,
    // per-layer logdet constants, then bf16 transposed weights.
    float* wsf     = (float*)d_ws;
    float* Hg      = wsf;                                   // B*D
    float* partial = wsf + (size_t)Bsz * Dd;                // NWG*512
    float* stats   = partial + (size_t)NWG * 512;           // 512
    float* scaleV  = stats + 512;                           // L*D
    float* ldconst = scaleV + (size_t)Lw * Dd;              // L (pad to 8)
    __bf16* W1t  = (__bf16*)(ldconst + 8);                  // L*256*128
    __bf16* W2t  = W1t + (size_t)Lw * Dd * S1c;             // L*256*256
    __bf16* Pt   = W2t + (size_t)Lw * Dd * Dd;              // L*256*256
    __bf16* Wh1t = Pt  + (size_t)Lw * Dd * Dd;              // 256*256

    const int convTotal = Lw * Dd * S1c + 2 * Lw * Dd * Dd + Dd * Dd;  // 1376256
    kConvert<<<(convTotal + 255) / 256, 256, 0, stream>>>(W1, W2, P, Wh1,
                                                          W1t, W2t, Pt, Wh1t);
    kScale<<<Lw, 256, 0, stream>>>(gs, scaleV, ldconst);
    kFlow<<<NWG, 256, 0, stream>>>(feat, b1, b2, go, bh1,
                                   W1t, W2t, Pt, Wh1t, scaleV, ldconst,
                                   Hg, partial, out);
    kStats<<<1, 256, 0, stream>>>(partial, gamma, beta, stats);
    kLogits<<<Bsz / 256, 256, 0, stream>>>(Hg, stats, Wh2, bh2, out);
}
